// ConditionalHandwritingRNN_87479893885184
// MI455X (gfx1250) — matrix operand order repaired and checked by reading, not compiled, not run on hardware
//
#include <hip/hip_runtime.h>
#include <hip/hip_bf16.h>

// ---------------------------------------------------------------------------
// Problem dims
// ---------------------------------------------------------------------------
#define BB   64
#define TT   800
#define UU   80
#define INN  3
#define HH   400
#define VV   57
#define KK   10
#define GG   20
#define G3   1200          // 3*H
#define ROWS (BB*TT)       // 51200
#define KP   416           // H padded to mult of 32 (13 k-tiles)
#define DP   480           // D1=460 padded to mult of 32 (15 k-tiles)
#define MDN_COLS 121       // 1 + 6*G

typedef __bf16 bf16_t;
typedef __attribute__((ext_vector_type(16))) __bf16        v16bf;
typedef __attribute__((ext_vector_type(8)))  float         v8f;
typedef __attribute__((ext_vector_type(4)))  unsigned int  u32x4;

union FragAB { v16bf v; u32x4 q[2]; };

// A-matrix (16x32, M x K) tile load per CDNA5 ISA layout:
// lane%16 = row M; lane-half h: K chunks [k0+8h, +8) and [k0+16+8h, +8).
__device__ __forceinline__ v16bf load_a_tile(const bf16_t* base, int ldk,
                                             int m0, int k0, int lane) {
    int row = m0 + (lane & 15);
    int h   = lane >> 4;
    const bf16_t* p = base + (size_t)row * ldk + k0 + 8 * h;
    FragAB f;
    f.q[0] = *(const u32x4*)(p);        // K = k0+8h .. +8
    f.q[1] = *(const u32x4*)(p + 16);   // K = k0+16+8h .. +8
    return f.v;
}

// B-matrix (32x16, K x N) tile from row-major (N,K) weights:
// lane%16 = column N; lane-half h: K chunks [k0+8h, +8) and [k0+16+8h, +8).
__device__ __forceinline__ v16bf load_b_tile(const bf16_t* W, int ldk,
                                             int n0, int k0, int lane) {
    int row = n0 + (lane & 15);
    int h   = lane >> 4;
    const bf16_t* p = W + (size_t)row * ldk + k0 + 8 * h;
    FragAB f;
    f.q[0] = *(const u32x4*)(p);
    f.q[1] = *(const u32x4*)(p + 16);
    return f.v;
}

__device__ __forceinline__ v8f wmma_bf16(v16bf a, v16bf b, v8f c) {
    // emits v_wmma_f32_16x16x32_bf16
    return __builtin_amdgcn_wmma_f32_16x16x32_bf16(
        /*neg_a=*/false, a, /*neg_b=*/false, b,
        /*c_mod=*/(short)0, c, /*reuse_a=*/false, /*reuse_b=*/false);
}

__device__ __forceinline__ float sigmoidf_(float v) {
    return 1.0f / (1.0f + __expf(-v));
}

// ---------------------------------------------------------------------------
// Weight convert + zero-pad  f32 (rs,cs) -> bf16 (rd,cd)
// ---------------------------------------------------------------------------
__global__ void cvt_pad_kernel(bf16_t* __restrict__ dst, const float* __restrict__ src,
                               int rs, int cs, int rd, int cd) {
    int idx = blockIdx.x * blockDim.x + threadIdx.x;
    if (idx >= rd * cd) return;
    int r = idx / cd, c = idx % cd;
    dst[idx] = (r < rs && c < cs) ? (bf16_t)src[r * cs + c] : (bf16_t)0.0f;
}

// ---------------------------------------------------------------------------
// Serial GRU scan: one 32-wave workgroup, h in LDS (bf16 ping-pong + f32),
// per-step h@W_hh^T via WMMA. mode 0: xg computed inline from x (D=3).
// mode 1: xg preloaded from global (already includes b_ih).
// Job = (m-tile, hidden-unit tile): computes r/z/n columns together so the
// gate nonlinearity stays in registers.  One barrier per step (ping-pong).
// ---------------------------------------------------------------------------
__global__ void __launch_bounds__(1024)
gru_scan_kernel(const bf16_t* __restrict__ Whh,    // (1200, KP) bf16, K-padded
                const float*  __restrict__ b_hh,   // (1200)
                const float*  __restrict__ h_init, // (64, 400)
                const float*  __restrict__ xg,     // (ROWS, 1200) f32, mode 1
                const float*  __restrict__ xin,    // (B,T,3) f32, mode 0
                const float*  __restrict__ Wih0,   // (1200, 3), mode 0
                const float*  __restrict__ b_ih0,  // (1200), mode 0
                bf16_t*       __restrict__ concat, // (ROWS, DP): write cols 0..400
                int mode) {
    extern __shared__ char smem[];
    bf16_t* hbf0 = (bf16_t*)smem;            // 64 x KP
    bf16_t* hbf1 = hbf0 + BB * KP;           // 64 x KP
    float*  hf   = (float*)(hbf1 + BB * KP); // 64 x 400

    const int tid  = threadIdx.x;
    const int lane = tid & 31;
    const int wid  = tid >> 5;

    for (int i = tid; i < BB * KP; i += 1024) {
        int r = i / KP, c = i % KP;
        float v = (c < HH) ? h_init[r * HH + c] : 0.0f;
        hbf0[i] = (bf16_t)v;
        hbf1[i] = (bf16_t)0.0f;
        if (c < HH) hf[r * HH + c] = v;
    }
    __syncthreads();

    for (int t = 0; t < TT; ++t) {
        bf16_t* cur = (t & 1) ? hbf1 : hbf0;
        bf16_t* nxt = (t & 1) ? hbf0 : hbf1;

        for (int job = wid; job < 100; job += 32) {
            const int mt = job & 3;        // 0..3  (batch tile)
            const int nt = job >> 2;       // 0..24 (hidden-unit tile)
            const int m0 = mt * 16, n0 = nt * 16;

            v8f ar = {}; v8f az = {}; v8f an = {};
            #pragma unroll
            for (int kt = 0; kt < KP / 32; ++kt) {
                const int k0 = kt * 32;
                v16bf a  = load_a_tile(cur, KP, m0, k0, lane);
                v16bf br = load_b_tile(Whh, KP, n0,          k0, lane);
                v16bf bz = load_b_tile(Whh, KP, HH + n0,     k0, lane);
                v16bf bn = load_b_tile(Whh, KP, 2 * HH + n0, k0, lane);
                ar = wmma_bf16(a, br, ar);
                az = wmma_bf16(a, bz, az);
                an = wmma_bf16(a, bn, an);
            }

            const int col = n0 + (lane & 15);   // hidden unit index
            const int hh  = lane >> 4;
            const float bhr = b_hh[col], bhz = b_hh[HH + col], bhn = b_hh[2 * HH + col];

            float bir = 0.f, biz = 0.f, bin_ = 0.f;
            float wr0 = 0.f, wr1 = 0.f, wr2 = 0.f;
            float wz0 = 0.f, wz1 = 0.f, wz2 = 0.f;
            float wn0 = 0.f, wn1 = 0.f, wn2 = 0.f;
            if (mode == 0) {
                bir = b_ih0[col]; biz = b_ih0[HH + col]; bin_ = b_ih0[2 * HH + col];
                wr0 = Wih0[col * 3 + 0];            wr1 = Wih0[col * 3 + 1];            wr2 = Wih0[col * 3 + 2];
                wz0 = Wih0[(HH + col) * 3 + 0];     wz1 = Wih0[(HH + col) * 3 + 1];     wz2 = Wih0[(HH + col) * 3 + 2];
                wn0 = Wih0[(2 * HH + col) * 3 + 0]; wn1 = Wih0[(2 * HH + col) * 3 + 1]; wn2 = Wih0[(2 * HH + col) * 3 + 2];
            }

            #pragma unroll
            for (int r = 0; r < 8; ++r) {
                const int b   = m0 + r + 8 * hh;   // batch index (C/D layout)
                const int row = b * TT + t;        // flat (b,t) row
                float xr, xz, xn;
                if (mode == 0) {
                    float x0 = xin[row * 3 + 0];
                    float x1 = xin[row * 3 + 1];
                    float x2 = xin[row * 3 + 2];
                    xr = bir  + wr0 * x0 + wr1 * x1 + wr2 * x2;
                    xz = biz  + wz0 * x0 + wz1 * x1 + wz2 * x2;
                    xn = bin_ + wn0 * x0 + wn1 * x1 + wn2 * x2;
                } else {
                    const float* xgr = xg + (size_t)row * G3;
                    xr = xgr[col]; xz = xgr[HH + col]; xn = xgr[2 * HH + col];
                }
                float ghr = ar[r] + bhr, ghz = az[r] + bhz, ghn = an[r] + bhn;
                float rr = sigmoidf_(xr + ghr);
                float zz = sigmoidf_(xz + ghz);
                float nn = tanhf(xn + rr * ghn);
                float hold = hf[b * HH + col];
                float hnew = (1.0f - zz) * nn + zz * hold;
                hf[b * HH + col]  = hnew;                 // owner-only slot
                nxt[b * KP + col] = (bf16_t)hnew;         // next step's A
                concat[(size_t)row * DP + col] = (bf16_t)hnew;  // out_t
            }
        }
        __syncthreads();   // ping-pong: single barrier per step
    }
}

// ---------------------------------------------------------------------------
// xg GEMM: (ROWS x DP bf16) x (1200 x DP bf16)^T + bias -> (ROWS x 1200) f32
// One 16x16 tile per wave, 15 k-tiles of 32.
// ---------------------------------------------------------------------------
__global__ void __launch_bounds__(256)
gemm_xg_kernel(const bf16_t* __restrict__ A, const bf16_t* __restrict__ W,
               const float* __restrict__ bias, float* __restrict__ C) {
    const int lane = threadIdx.x & 31;
    const int wave = blockIdx.x * (blockDim.x >> 5) + (threadIdx.x >> 5);
    const int Mtiles = ROWS / 16;   // 3200
    const int Ntiles = G3 / 16;     // 75
    if (wave >= Mtiles * Ntiles) return;
    const int mt = wave / Ntiles, nt = wave % Ntiles;
    const int m0 = mt * 16, n0 = nt * 16;

    v8f acc = {};
    #pragma unroll
    for (int kt = 0; kt < DP / 32; ++kt) {
        const int k0 = kt * 32;
        v16bf a = load_a_tile(A, DP, m0, k0, lane);
        v16bf b = load_b_tile(W, DP, n0, k0, lane);
        if (kt + 1 < DP / 32)   // global_prefetch_b8 of next B k-tile
            __builtin_prefetch(W + (size_t)(n0 + (lane & 15)) * DP + k0 + 32, 0, 0);
        acc = wmma_bf16(a, b, acc);
    }
    const int col = n0 + (lane & 15);
    const int hh  = lane >> 4;
    const float bv = bias[col];
    #pragma unroll
    for (int r = 0; r < 8; ++r) {
        const int row = m0 + r + 8 * hh;
        C[(size_t)row * G3 + col] = acc[r] + bv;
    }
}

// ---------------------------------------------------------------------------
// MDN head: concat cols 0..416 (pad rows of W_mdn_bf are zero) x (128 x 416)
// epilogue: +bias, tanh on cols 100..119, write 121 cols per row.
// ---------------------------------------------------------------------------
__global__ void __launch_bounds__(256)
mdn_kernel(const bf16_t* __restrict__ A, const bf16_t* __restrict__ W,
           const float* __restrict__ bias, float* __restrict__ out) {
    const int lane = threadIdx.x & 31;
    const int wave = blockIdx.x * (blockDim.x >> 5) + (threadIdx.x >> 5);
    const int Mtiles = ROWS / 16;   // 3200
    const int Ntiles = 128 / 16;    // 8
    if (wave >= Mtiles * Ntiles) return;
    const int mt = wave / Ntiles, nt = wave % Ntiles;
    const int m0 = mt * 16, n0 = nt * 16;

    v8f acc = {};
    #pragma unroll
    for (int kt = 0; kt < KP / 32; ++kt) {
        const int k0 = kt * 32;
        v16bf a = load_a_tile(A, DP, m0, k0, lane);   // lda = DP, use first 416
        v16bf b = load_b_tile(W, KP, n0, k0, lane);
        acc = wmma_bf16(a, b, acc);
    }
    const int col = n0 + (lane & 15);
    if (col >= MDN_COLS) return;
    const int hh = lane >> 4;
    const float bv = bias[col];
    #pragma unroll
    for (int r = 0; r < 8; ++r) {
        const int row = m0 + r + 8 * hh;
        float v = acc[r] + bv;
        if (col >= 5 * GG && col < 6 * GG) v = tanhf(v);   // rho
        out[(size_t)row * MDN_COLS + col] = v;
    }
}

// ---------------------------------------------------------------------------
// Attention: abk = exp(out0 @ W_win^T + b_win)   (ROWS x 30)
// ---------------------------------------------------------------------------
__global__ void abk_kernel(const bf16_t* __restrict__ concat,
                           const float* __restrict__ Wwin,
                           const float* __restrict__ bwin,
                           float* __restrict__ abk) {
    int idx = blockIdx.x * blockDim.x + threadIdx.x;
    if (idx >= ROWS * 3 * KK) return;
    int row = idx / (3 * KK), j = idx % (3 * KK);
    const bf16_t* a = concat + (size_t)row * DP;
    const float*  w = Wwin + j * HH;
    float s = bwin[j];
    for (int k = 0; k < HH; ++k) s += (float)a[k] * w[k];
    abk[idx] = __expf(s);
}

// cumsum of kappa (third chunk of abk) over t, per (b,k): 640 serial chains
__global__ void cumsum_kernel(float* __restrict__ abk) {
    int idx = blockIdx.x * blockDim.x + threadIdx.x;
    if (idx >= BB * KK) return;
    int b = idx / KK, k = idx % KK;
    float run = 0.0f;
    for (int t = 0; t < TT; ++t) {
        size_t p = ((size_t)(b * TT + t)) * (3 * KK) + 2 * KK + k;
        run += abk[p];
        abk[p] = run;
    }
}

// phi + window; also writes x and zero-pad into concat cols 400..480
__global__ void __launch_bounds__(128)
window_kernel(const float* __restrict__ abk, const float* __restrict__ onehot,
              const float* __restrict__ xin, bf16_t* __restrict__ concat) {
    __shared__ float sabk[3 * KK];
    __shared__ float phi[UU];
    const int row = blockIdx.x;       // b*T + t
    const int b   = row / TT;
    const int tid = threadIdx.x;

    if (tid < 3 * KK) sabk[tid] = abk[(size_t)row * (3 * KK) + tid];
    __syncthreads();

    if (tid < UU) {
        float u = (float)tid, s = 0.0f;
        #pragma unroll
        for (int k = 0; k < KK; ++k) {
            float d = sabk[2 * KK + k] - u;            // kappa - u
            s += sabk[k] * __expf(-sabk[KK + k] * d * d); // alpha*exp(-beta*d^2)
        }
        phi[tid] = s;
    }
    __syncthreads();

    bf16_t* crow = concat + (size_t)row * DP;
    if (tid < VV) {
        float s = 0.0f;
        for (int u = 0; u < UU; ++u)
            s += phi[u] * onehot[((size_t)b * UU + u) * VV + tid];
        crow[HH + tid] = (bf16_t)s;                        // cols 400..457
    } else if (tid < VV + INN) {
        crow[HH + tid] = (bf16_t)xin[(size_t)row * INN + (tid - VV)]; // 457..460
    } else if (tid < 80) {
        crow[HH + VV + INN + (tid - VV - INN)] = (bf16_t)0.0f;        // 460..480
    }
}

// ---------------------------------------------------------------------------
// Host launch
// ---------------------------------------------------------------------------
extern "C" void kernel_launch(void* const* d_in, const int* in_sizes, int n_in,
                              void* d_out, int out_size, void* d_ws, size_t ws_size,
                              hipStream_t stream) {
    const float* x      = (const float*)d_in[0];
    const float* onehot = (const float*)d_in[1];
    const float* h0     = (const float*)d_in[2];
    const float* h1     = (const float*)d_in[3];
    const float* h2     = (const float*)d_in[4];
    const float* W_ih0  = (const float*)d_in[5];
    const float* W_hh0  = (const float*)d_in[6];
    const float* b_ih0  = (const float*)d_in[7];
    const float* b_hh0  = (const float*)d_in[8];
    const float* W_ih1  = (const float*)d_in[9];
    const float* W_hh1  = (const float*)d_in[10];
    const float* b_ih1  = (const float*)d_in[11];
    const float* b_hh1  = (const float*)d_in[12];
    const float* W_win  = (const float*)d_in[13];
    const float* b_win  = (const float*)d_in[14];
    const float* W_mdn  = (const float*)d_in[15];
    const float* b_mdn  = (const float*)d_in[16];
    float* out = (float*)d_out;

    char* ws = (char*)d_ws;
    size_t off = 0;
    auto alloc = [&](size_t bytes) -> char* {
        char* p = ws + off;
        off += (bytes + 255) & ~(size_t)255;
        return p;
    };
    bf16_t* Whh0_bf = (bf16_t*)alloc((size_t)G3 * KP * 2);
    bf16_t* Whh1_bf = (bf16_t*)alloc((size_t)G3 * KP * 2);
    bf16_t* Wih1_bf = (bf16_t*)alloc((size_t)G3 * DP * 2);
    bf16_t* Wmdn_bf = (bf16_t*)alloc((size_t)128 * KP * 2);
    float*  abk     = (float*) alloc((size_t)ROWS * 3 * KK * 4);
    bf16_t* concat  = (bf16_t*)alloc((size_t)ROWS * DP * 4 / 2);   // ROWS*DP bf16
    float*  xg      = (float*) alloc((size_t)ROWS * G3 * 4);

    // 1) bf16 weight copies, zero K-padded (zero B-pad makes A-pad irrelevant)
    cvt_pad_kernel<<<(G3 * KP + 255) / 256, 256, 0, stream>>>(Whh0_bf, W_hh0, G3, HH, G3, KP);
    cvt_pad_kernel<<<(G3 * KP + 255) / 256, 256, 0, stream>>>(Whh1_bf, W_hh1, G3, HH, G3, KP);
    cvt_pad_kernel<<<(G3 * DP + 255) / 256, 256, 0, stream>>>(Wih1_bf, W_ih1, G3, HH + INN + VV, G3, DP);
    cvt_pad_kernel<<<(128 * KP + 255) / 256, 256, 0, stream>>>(Wmdn_bf, W_mdn, MDN_COLS, HH, 128, KP);

    const size_t scan_lds = (size_t)BB * KP * 2 * 2 + (size_t)BB * HH * 4; // 208896 B

    // 2) GRU layer 0 (xg inline, D=3); out0 -> concat cols 0..400
    gru_scan_kernel<<<1, 1024, scan_lds, stream>>>(
        Whh0_bf, b_hh0, h0, nullptr, x, W_ih0, b_ih0, concat, 0);

    // 3) attention window -> concat cols 400..480 (+x, +zero pad)
    abk_kernel<<<(ROWS * 3 * KK + 255) / 256, 256, 0, stream>>>(concat, W_win, b_win, abk);
    cumsum_kernel<<<(BB * KK + 255) / 256, 256, 0, stream>>>(abk);
    window_kernel<<<ROWS, 128, 0, stream>>>(abk, onehot, x, concat);

    // 4) xg1 = concat @ W_ih1^T + b_ih1 ; GRU layer 1 -> concat cols 0..400
    gemm_xg_kernel<<<(ROWS / 16) * (G3 / 16) / 8, 256, 0, stream>>>(concat, Wih1_bf, b_ih1, xg);
    gru_scan_kernel<<<1, 1024, scan_lds, stream>>>(
        Whh1_bf, b_hh1, h1, xg, nullptr, nullptr, nullptr, concat, 1);

    // 5) xg2 (same weights, concat now holds out1) ; GRU layer 2 -> concat
    gemm_xg_kernel<<<(ROWS / 16) * (G3 / 16) / 8, 256, 0, stream>>>(concat, Wih1_bf, b_ih1, xg);
    gru_scan_kernel<<<1, 1024, scan_lds, stream>>>(
        Whh1_bf, b_hh1, h2, xg, nullptr, nullptr, nullptr, concat, 1);

    // 6) MDN head + epilogue -> d_out (ROWS x 121 f32)
    mdn_kernel<<<(ROWS / 16) * (128 / 16) / 8, 256, 0, stream>>>(concat, Wmdn_bf, b_mdn, out);

    (void)in_sizes; (void)n_in; (void)out_size; (void)ws_size;
}
